// HiFiGAT_53094385713961
// MI455X (gfx1250) — compile-verified, matrix operand order checked by reading, not run on hardware
//
#include <hip/hip_runtime.h>
#include <hip/hip_bf16.h>
#include <math.h>

// ---- problem constants (match reference) ----
#define N_NODES 50000
#define N_EDGES 800000
#define ETOT    (N_EDGES + N_NODES)   // self-loops appended
#define F_INC   128
#define HID     64
#define HEADS   4
#define NGRP    64
#define NCLS    10
#define SLOPE   0.2f

typedef __attribute__((ext_vector_type(16))) __bf16 v16bf;
typedef __attribute__((ext_vector_type(2)))  __bf16 v2bf;
typedef __attribute__((ext_vector_type(8)))  float  v8f;

// ---- order-preserving float<->uint encoding for atomic max ----
__device__ __forceinline__ unsigned fenc(float f) {
    unsigned u = __float_as_uint(f);
    return (u & 0x80000000u) ? ~u : (u | 0x80000000u);
}
__device__ __forceinline__ float fdec(unsigned u) {
    unsigned b = (u & 0x80000000u) ? (u & 0x7FFFFFFFu) : ~u;
    return __uint_as_float(b);
}
__device__ __forceinline__ bool finitef(float f) {
    return (__float_as_uint(f) & 0x7F800000u) != 0x7F800000u;
}
__device__ __forceinline__ float lrelu(float f) { return f > 0.f ? f : SLOPE * f; }

// pack two floats into two bf16 in one dword (-> v_cvt_pk_bf16_f32)
__device__ __forceinline__ unsigned pk_bf16(float lo, float hi) {
    v2bf p; p[0] = (__bf16)lo; p[1] = (__bf16)hi;
    return __builtin_bit_cast(unsigned, p);
}

// ============================================================
// bf16 WMMA GEMM, exact tiling: C[M,N] = act(A[M,K] @ B[K,N] + bias)
// Block = 128 threads (4 waves) -> one 16(M) x 64(N) output slab.
// K, N compile-time (all dims here are fixed): every staging load gets a
// compile-time immediate offset from a single base pointer -> one clause,
// one wait, no per-load address math.
// A staged as [16 m][16 k-pairs] packed bf16x2; B staged transposed
// [64 n][16 k-pairs] so each lane's fragment is 4x ds_load_b128.
// ACT: 0=none 1=relu 2=leaky(0.2)
// ============================================================
template<int K, int N, int ACT>
__global__ __launch_bounds__(128)
void gemm_wmma_bf16(const float* __restrict__ A, const float* __restrict__ B,
                    const float* __restrict__ bias, float* __restrict__ C, int M)
{
    __shared__ unsigned lA[16][16];
    __shared__ unsigned lB[64][16];
    constexpr int slabsN = N >> 6;
    const int tm   = blockIdx.x / slabsN;
    const int tn0  = (blockIdx.x % slabsN) << 6;
    const int tid  = threadIdx.x;            // always < 128
    const int wave = tid >> 5;
    const int lane = tid & 31;
    const int half = lane >> 4;
    const int mn   = lane & 15;

    // A staging: elements tid and tid+128 of the 16x16 pair grid
    const int am0 = tid >> 4, am1 = (tid + 128) >> 4, apx = tid & 15;
    const float* abase0 = A + (unsigned)((tm * 16 + am0) * K + 2 * apx);
    const float* abase1 = A + (unsigned)((tm * 16 + am1) * K + 2 * apx);
    // B staging: elements tid + 128*r of the 64x16 pair grid -> n=tid&63, p=bp0+2r
    const int bn  = tid & 63;
    const int bp0 = tid >> 6;
    const float* bbase = B + (unsigned)(2 * bp0 * N + tn0 + bn);
    const float bv = bias ? bias[tn0 + wave * 16 + mn] : 0.f;

    v8f acc = {};
#pragma unroll
    for (int k0 = 0; k0 < K; k0 += 32) {
        // ---- batch all global loads (compile-time offsets -> one clause) ----
        float2 a0 = *(const float2*)(abase0 + k0);
        float2 a1 = *(const float2*)(abase1 + k0);
        float blo[8], bhi[8];
        const float* bk = bbase + k0 * N;
#pragma unroll
        for (int r = 0; r < 8; ++r) {
            blo[r] = bk[4 * r * N];
            bhi[r] = bk[4 * r * N + N];
        }
        lA[am0][apx] = pk_bf16(a0.x, a0.y);
        lA[am1][apx] = pk_bf16(a1.x, a1.y);
#pragma unroll
        for (int r = 0; r < 8; ++r) lB[bn][bp0 + 2 * r] = pk_bf16(blo[r], bhi[r]);
        __syncthreads();
        // ---- build fragments (pairs of consecutive K = one dword) ----
        v16bf a, b;
#pragma unroll
        for (int j = 0; j < 8; ++j) {
            int p = ((j < 4) ? 0 : 8) + (half ? 4 : 0) + (j & 3);
            v2bf pa = __builtin_bit_cast(v2bf, lA[mn][p]);
            v2bf pb = __builtin_bit_cast(v2bf, lB[wave * 16 + mn][p]);
            a[2 * j] = pa[0]; a[2 * j + 1] = pa[1];
            b[2 * j] = pb[0]; b[2 * j + 1] = pb[1];
        }
        acc = __builtin_amdgcn_wmma_f32_16x16x32_bf16(false, a, false, b,
                                                      (short)0, acc, false, false);
        __syncthreads();
    }
    const int ncol = tn0 + wave * 16 + mn;
#pragma unroll
    for (int r = 0; r < 8; ++r) {
        int row = tm * 16 + r + (half ? 8 : 0);
        float v = acc[r] + bv;
        if (ACT == 1) v = v > 0.f ? v : 0.f;
        else if (ACT == 2) v = lrelu(v);
        C[(unsigned)(row * N + ncol)] = v;
    }
}

// ---- attention scores: one wave per (n,h); lanes over channels, shfl reduce ----
template<int H>
__global__ void attn_scores(const float* __restrict__ ht, const float* __restrict__ att_s,
                            const float* __restrict__ att_d, float* __restrict__ as,
                            float* __restrict__ ad)
{
    int wv = blockIdx.x * (blockDim.x >> 5) + (threadIdx.x >> 5);
    if (wv >= N_NODES * H) return;
    int h = wv % H;
    int lane = threadIdx.x & 31;
    const float* hp = ht + (size_t)wv * HID;            // ht is [N, H, HID] contiguous
    float v0 = hp[lane], v1 = hp[lane + 32];
    float s = v0 * att_s[h * HID + lane] + v1 * att_s[h * HID + lane + 32];
    float d = v0 * att_d[h * HID + lane] + v1 * att_d[h * HID + lane + 32];
#pragma unroll
    for (int o = 16; o; o >>= 1) { s += __shfl_xor(s, o, 32); d += __shfl_xor(d, o, 32); }
    if (lane == 0) { as[wv] = s; ad[wv] = d; }
}

__device__ __forceinline__ void edge_sd(const int* __restrict__ ei, int e, int& s, int& d) {
    if (e < N_EDGES) { s = ei[e]; d = ei[N_EDGES + e]; }
    else             { s = d = e - N_EDGES; }            // self-loops
}

// ---- pass 1: per-destination segment max of leaky(a_src+a_dst) ----
template<int H>
__global__ void edge_max(const int* __restrict__ ei, const float* __restrict__ as,
                         const float* __restrict__ ad, unsigned* __restrict__ amax)
{
    int idx = blockIdx.x * blockDim.x + threadIdx.x;
    if (idx >= ETOT * H) return;
    int e = idx / H, h = idx % H;
    int s, d; edge_sd(ei, e, s, d);
    float alpha = lrelu(as[s * H + h] + ad[d * H + h]);
    atomicMax(&amax[d * H + h], fenc(alpha));
}

// decode encoded max back to float; non-finite -> 0 (matches reference)
__global__ void decode_max(float* buf, int n)
{
    int i = blockIdx.x * blockDim.x + threadIdx.x;
    if (i >= n) return;
    float f = fdec(((unsigned*)buf)[i]);
    buf[i] = finitef(f) ? f : 0.f;
}

// ---- pass 2: denom = segment_sum(exp(alpha - amax[dst])) ----
template<int H>
__global__ void edge_sum(const int* __restrict__ ei, const float* __restrict__ as,
                         const float* __restrict__ ad, const float* __restrict__ amax,
                         float* __restrict__ den)
{
    int idx = blockIdx.x * blockDim.x + threadIdx.x;
    if (idx >= ETOT * H) return;
    int e = idx / H, h = idx % H;
    int s, d; edge_sd(ei, e, s, d);
    float alpha = lrelu(as[s * H + h] + ad[d * H + h]);
    atomicAdd(&den[d * H + h], expf(alpha - amax[d * H + h]));
}

// ---- pass 3: out[dst] += ht[src] * softmax weight ; one block per edge ----
template<int H>
__global__ void edge_msg(const int* __restrict__ ei, const float* __restrict__ as,
                         const float* __restrict__ ad, const float* __restrict__ amax,
                         const float* __restrict__ den, const float* __restrict__ ht,
                         float* __restrict__ out)
{
    int e = blockIdx.x;
    int tid = threadIdx.x;          // H*HID threads
    int h = tid >> 6;               // HID == 64
    int s, d; edge_sd(ei, e, s, d);
    float alpha = lrelu(as[s * H + h] + ad[d * H + h]);
    float w = expf(alpha - amax[d * H + h]) / (den[d * H + h] + 1e-16f);
    atomicAdd(&out[(size_t)d * H * HID + tid], ht[(size_t)s * H * HID + tid] * w);
}

// ---- out = leaky(out + bias[col]) in place; cols is a power of two ----
__global__ void bias_leaky(float* __restrict__ buf, const float* __restrict__ bias,
                           int n, int colmask)
{
    int i = blockIdx.x * blockDim.x + threadIdx.x;
    if (i >= n) return;
    buf[i] = lrelu(buf[i] + bias[i & colmask]);
}

// ---- graph pooling: mean-sum, max, counts ----
__global__ void pool(const float* __restrict__ h2, const int* __restrict__ batch,
                     float* __restrict__ gsum, unsigned* __restrict__ gmax,
                     float* __restrict__ cnt)
{
    int i = blockIdx.x * blockDim.x + threadIdx.x;
    if (i >= N_NODES * HID) return;
    int n = i >> 6, c = i & 63;
    int g = batch[n];
    float v = h2[i];
    atomicAdd(&gsum[g * HID + c], v);
    atomicMax(&gmax[g * HID + c], fenc(v));
    if (c == 0) atomicAdd(&cnt[g], 1.f);
}

// ---- tiny MLP head + log_softmax : one thread per group row ----
__global__ void head(const float* __restrict__ gsum, const unsigned* __restrict__ gmaxe,
                     const float* __restrict__ cnt, const float* __restrict__ Wf1,
                     const float* __restrict__ bf1, const float* __restrict__ Wf2,
                     const float* __restrict__ bf2, float* __restrict__ out)
{
    int t = threadIdx.x;
    if (t >= NGRP) return;
    float gv[HID];
    float inv = 1.f / fmaxf(cnt[t], 1.f);
    for (int c = 0; c < HID; ++c) {
        float m = fdec(gmaxe[t * HID + c]);
        if (!finitef(m)) m = 0.f;
        gv[c] = gsum[t * HID + c] * inv + m;
    }
    float f1[HID / 2];
    for (int j = 0; j < HID / 2; ++j) {
        float s = bf1[j];
        for (int c = 0; c < HID; ++c) s += gv[c] * Wf1[c * (HID / 2) + j];
        f1[j] = s > 0.f ? s : 0.f;
    }
    float o[NCLS]; float mx = -1e30f;
    for (int j = 0; j < NCLS; ++j) {
        float s = bf2[j];
        for (int c = 0; c < HID / 2; ++c) s += f1[c] * Wf2[c * NCLS + j];
        o[j] = s; mx = fmaxf(mx, s);
    }
    float lse = 0.f;
    for (int j = 0; j < NCLS; ++j) lse += expf(o[j] - mx);
    lse = logf(lse);
    for (int j = 0; j < NCLS; ++j) out[t * NCLS + j] = o[j] - mx - lse;
}

// ============================================================
extern "C" void kernel_launch(void* const* d_in, const int* in_sizes, int n_in,
                              void* d_out, int out_size, void* d_ws, size_t ws_size,
                              hipStream_t stream)
{
    (void)in_sizes; (void)n_in; (void)out_size; (void)ws_size;
    const float* x        = (const float*)d_in[0];
    const int*   ei       = (const int*)  d_in[1];
    const int*   batch    = (const int*)  d_in[2];
    const float* W_proj   = (const float*)d_in[3];
    const float* b_proj   = (const float*)d_in[4];
    const float* W1       = (const float*)d_in[5];
    const float* att_s1   = (const float*)d_in[6];
    const float* att_d1   = (const float*)d_in[7];
    const float* b1       = (const float*)d_in[8];
    const float* W2       = (const float*)d_in[9];
    const float* att_s2   = (const float*)d_in[10];
    const float* att_d2   = (const float*)d_in[11];
    const float* b2       = (const float*)d_in[12];
    const float* W_fc1    = (const float*)d_in[13];
    const float* b_fc1    = (const float*)d_in[14];
    const float* W_fc2    = (const float*)d_in[15];
    const float* b_fc2    = (const float*)d_in[16];
    float* out = (float*)d_out;

    // ---- workspace layout (floats) ----
    float* ws = (float*)d_ws;
    const size_t OFF_H0   = 0;                                   // N*64  (later ht2)
    const size_t OFF_HT1  = (size_t)N_NODES * HID;               // N*256 (later out2)
    const size_t OFF_OUT1 = OFF_HT1 + (size_t)N_NODES * 256;     // N*256 (becomes h1)
    const size_t S        = OFF_OUT1 + (size_t)N_NODES * 256;
    const size_t OFF_AS1  = S;
    const size_t OFF_AD1  = OFF_AS1  + (size_t)N_NODES * HEADS;
    const size_t OFF_AM1  = OFF_AD1  + (size_t)N_NODES * HEADS;
    const size_t OFF_DN1  = OFF_AM1  + (size_t)N_NODES * HEADS;
    const size_t OFF_AS2  = OFF_DN1  + (size_t)N_NODES * HEADS;
    const size_t OFF_AD2  = OFF_AS2  + (size_t)N_NODES;
    const size_t OFF_AM2  = OFF_AD2  + (size_t)N_NODES;
    const size_t OFF_DN2  = OFF_AM2  + (size_t)N_NODES;
    const size_t OFF_GSUM = OFF_DN2  + (size_t)N_NODES;
    const size_t OFF_GMAX = OFF_GSUM + (size_t)NGRP * HID;
    const size_t OFF_CNT  = OFF_GMAX + (size_t)NGRP * HID;
    const size_t TOTAL    = OFF_CNT + NGRP;

    float* h0   = ws + OFF_H0;     // then ht2
    float* ht1  = ws + OFF_HT1;    // then out2
    float* out1 = ws + OFF_OUT1;   // then h1
    float* as1 = ws + OFF_AS1; float* ad1 = ws + OFF_AD1;
    float* am1 = ws + OFF_AM1; float* dn1 = ws + OFF_DN1;
    float* as2 = ws + OFF_AS2; float* ad2 = ws + OFF_AD2;
    float* am2 = ws + OFF_AM2; float* dn2 = ws + OFF_DN2;
    float* gsum = ws + OFF_GSUM; float* gmax = ws + OFF_GMAX; float* cnt = ws + OFF_CNT;

    // zero out1 + all accumulator/stat buffers (raw 0 is the minimum fenc code -> valid init for max bufs)
    hipMemsetAsync(out1, 0, (TOTAL - OFF_OUT1) * sizeof(float), stream);

    const int TPB = 256;
    // 1) h0 = relu(x @ W_proj + b_proj)            [50000,128]x[128,64]
    gemm_wmma_bf16<F_INC, HID, 1><<<(N_NODES / 16) * (HID / 64), 128, 0, stream>>>(
        x, W_proj, b_proj, h0, N_NODES);
    // 2) ht1 = h0 @ W1                             [50000,64]x[64,256]
    gemm_wmma_bf16<HID, HEADS * HID, 0><<<(N_NODES / 16) * (256 / 64), 128, 0, stream>>>(
        h0, W1, nullptr, ht1, N_NODES);
    // 3) attention logits, edge softmax (H=4)
    attn_scores<HEADS><<<(N_NODES * HEADS + 7) / 8, TPB, 0, stream>>>(ht1, att_s1, att_d1,
                                                                      as1, ad1);
    edge_max<HEADS><<<(ETOT * HEADS + TPB - 1) / TPB, TPB, 0, stream>>>(ei, as1, ad1,
                                                                        (unsigned*)am1);
    decode_max<<<(N_NODES * HEADS + TPB - 1) / TPB, TPB, 0, stream>>>(am1, N_NODES * HEADS);
    edge_sum<HEADS><<<(ETOT * HEADS + TPB - 1) / TPB, TPB, 0, stream>>>(ei, as1, ad1,
                                                                        am1, dn1);
    edge_msg<HEADS><<<ETOT, HEADS * HID, 0, stream>>>(ei, as1, ad1, am1, dn1, ht1, out1);
    // 4) h1 = leaky(out1 + b1)  (in place)
    bias_leaky<<<(N_NODES * 256 + TPB - 1) / TPB, TPB, 0, stream>>>(out1, b1,
                                                                    N_NODES * 256, 255);
    // 5) ht2 = h1 @ W2                             [50000,256]x[256,64]  (reuses h0 region)
    gemm_wmma_bf16<HEADS * HID, HID, 0><<<(N_NODES / 16) * (HID / 64), 128, 0, stream>>>(
        out1, W2, nullptr, h0, N_NODES);
    float* ht2  = h0;
    float* out2 = ht1;  // ht1 is dead now; reuse first N*64 floats
    hipMemsetAsync(out2, 0, (size_t)N_NODES * HID * sizeof(float), stream);
    // 6) edge softmax (H=1) + aggregation
    attn_scores<1><<<(N_NODES + 7) / 8, TPB, 0, stream>>>(ht2, att_s2, att_d2, as2, ad2);
    edge_max<1><<<(ETOT + TPB - 1) / TPB, TPB, 0, stream>>>(ei, as2, ad2, (unsigned*)am2);
    decode_max<<<(N_NODES + TPB - 1) / TPB, TPB, 0, stream>>>(am2, N_NODES);
    edge_sum<1><<<(ETOT + TPB - 1) / TPB, TPB, 0, stream>>>(ei, as2, ad2, am2, dn2);
    edge_msg<1><<<ETOT, HID, 0, stream>>>(ei, as2, ad2, am2, dn2, ht2, out2);
    // 7) h2 = leaky(out2 + b2)
    bias_leaky<<<(N_NODES * HID + TPB - 1) / TPB, TPB, 0, stream>>>(out2, b2,
                                                                    N_NODES * HID, 63);
    // 8) pooling (mean + max per group) and MLP head
    pool<<<(N_NODES * HID + TPB - 1) / TPB, TPB, 0, stream>>>(out2, batch, gsum,
                                                              (unsigned*)gmax, cnt);
    head<<<1, NGRP, 0, stream>>>(gsum, (const unsigned*)gmax, cnt,
                                 W_fc1, b_fc1, W_fc2, b_fc2, out);
}
